// Projector_37452114821260
// MI455X (gfx1250) — matrix operand order repaired and checked by reading, not compiled, not run on hardware
//
#include <hip/hip_runtime.h>

// ---------------------------------------------------------------------------
// CDNA5 (gfx1250) attention-projector pipeline.
// All GEMMs via v_wmma_f32_16x16x32_bf16; B tiles staged into LDS with the
// Tensor Data Mover (tensor_load_to_lds, TENSORcnt-synchronized, double
// buffered), B fragments software-pipelined out of LDS so ds latency hides
// under WMMA execution.  B=16, N=M=1024, C=512.
// ---------------------------------------------------------------------------

typedef __attribute__((ext_vector_type(16))) __bf16 v16bf;
typedef __attribute__((ext_vector_type(8)))  __bf16 v8bf;
typedef __attribute__((ext_vector_type(8)))  float  v8f;
typedef unsigned int v4u __attribute__((ext_vector_type(4)));
typedef int          v8i __attribute__((ext_vector_type(8)));
typedef int          v4i __attribute__((ext_vector_type(4)));

#define LOG2E 1.4426950408889634f

// LDS tile: 16 rows x 512 bf16, row stride 528 elems (1056B) -> pad 32B/row.
#define LDS_ROW 528
#define LDS_TILE (16 * LDS_ROW)   // elems per buffer

#if __has_builtin(__builtin_amdgcn_tensor_load_to_lds)
#define USE_TDM 1
#else
#define USE_TDM 0
#endif

__device__ __forceinline__ v8f wmma_bf16(v16bf a, v16bf b, v8f c) {
  return __builtin_amdgcn_wmma_f32_16x16x32_bf16(
      /*neg_a=*/false, a, /*neg_b=*/false, b,
      /*c_mod=*/(short)0, c, /*reuse_a=*/false, /*reuse_b=*/false);
}

// A-operand 16x32 bf16 fragment: lane L (row=L&15, h=L>>4), off = kb*32 + h*8,
// holds K = kb*32 + h*16 .. +15 (contiguous).
__device__ __forceinline__ v16bf load_a_frag(const __bf16* rowp, int off) {
  v8bf lo = *(const v8bf*)(rowp + off + (off & 8));
  v8bf hi = *(const v8bf*)(rowp + off + (off & 8) + 8);
  return __builtin_shufflevector(lo, hi, 0, 1, 2, 3, 4, 5, 6, 7,
                                 8, 9, 10, 11, 12, 13, 14, 15);
}

// Software-pipelined 16-step K loop from LDS: fragment kb+1 is issued before
// WMMA kb so the ds wait lands after a WMMA, not before it.
#define K16_PIPELINED(ACC, AV, BROW)                                \
  do {                                                              \
    v16bf bv_ = *(const v16bf*)(BROW);                              \
    _Pragma("unroll")                                               \
    for (int kb_ = 0; kb_ < 15; ++kb_) {                            \
      v16bf bn_ = *(const v16bf*)((BROW) + (kb_ + 1) * 32);         \
      (ACC) = wmma_bf16((AV)[kb_], bv_, (ACC));                     \
      bv_ = bn_;                                                    \
    }                                                               \
    (ACC) = wmma_bf16((AV)[15], bv_, (ACC));                        \
  } while (0)

// ---------------------------------------------------------------------------
// TDM stage: DMA a 16-row x 1KB 2-D tile (row stride rowStrideElems bf16)
// from global into LDS at `lds`, padding 32B after each 1KB row.
// ---------------------------------------------------------------------------
__device__ __forceinline__ void stage_issue(const __bf16* g, int rowStrideElems,
                                            __bf16* lds, int tid) {
#if USE_TDM
  if (tid < 32) {
    unsigned long long ga = (unsigned long long)(uintptr_t)g;
    unsigned int ldsa = (unsigned int)(uintptr_t)lds;  // generic low 32 = LDS offset
    v4u g0;
    g0.x = 0x1u;                                   // count=1 (valid descriptor)
    g0.y = ldsa;                                   // lds_addr
    g0.z = (unsigned int)ga;                       // global_addr[31:0]
    g0.w = (unsigned int)(ga >> 32) | (2u << 30);  // global_addr[56:32] | type=2
    v8i g1;
    // data_size=3 (8B units) | pad_enable | pad_interval=7 (128x8B=1KB)
    // | pad_amount=7 (8 dwords = 32B)
    g1[0] = (3 << 16) | (1 << 20) | (7 << 22) | (7 << 25);
    g1[1] = (int)(128u << 16);   // tensor_dim0[15:0]=128 (8B units per row)
    g1[2] = (int)(16u << 16);    // tensor_dim0 hi=0 | tensor_dim1[15:0]=16
    g1[3] = (int)(128u << 16);   // tensor_dim1 hi=0 | tile_dim0=128
    g1[4] = 16;                  // tile_dim1=16 | tile_dim2=0
    g1[5] = rowStrideElems >> 2; // tensor_dim0_stride[31:0] (8B units)
    g1[6] = 0;
    g1[7] = 0;
    v4i z4 = {0, 0, 0, 0};
    v8i z8 = {0, 0, 0, 0, 0, 0, 0, 0};
    __builtin_amdgcn_tensor_load_to_lds(g0, g1, z4, z4, z8, 0);
  }
#else
  // Cooperative fallback (blockDim==256): 16B per thread per iteration.
  for (int i = tid; i < 1024; i += 256) {
    int j = i << 3;
    int row = j >> 9, col = j & 511;
    *(v8bf*)(lds + row * LDS_ROW + col) =
        *(const v8bf*)(g + (size_t)row * rowStrideElems + col);
  }
#endif
}

__device__ __forceinline__ void stage_wait_pending(int tid) {
#if USE_TDM
  if (tid < 32) __builtin_amdgcn_s_wait_tensorcnt(1);
#endif
}
__device__ __forceinline__ void stage_wait_last(int tid) {
#if USE_TDM
  if (tid < 32) __builtin_amdgcn_s_wait_tensorcnt(0);
#endif
}

// ---------------------------------------------------------------------------
// 1) f32 -> bf16 elementwise convert (4 elems/thread)
// ---------------------------------------------------------------------------
__global__ void cvt_bf16_kernel(const float* __restrict__ in,
                                __bf16* __restrict__ out, int n) {
  int i = (blockIdx.x * blockDim.x + threadIdx.x) * 4;
  if (i < n) {
    float4 f = *(const float4*)(in + i);
    out[i + 0] = (__bf16)f.x;
    out[i + 1] = (__bf16)f.y;
    out[i + 2] = (__bf16)f.z;
    out[i + 3] = (__bf16)f.w;
  }
}

// ---------------------------------------------------------------------------
// 2) f32 [R,S] -> bf16 [S,R] batched transpose (32x32 LDS tile)
// ---------------------------------------------------------------------------
__global__ void transpose_cvt_kernel(const float* __restrict__ in,
                                     __bf16* __restrict__ out, int R, int S) {
  __shared__ float tile[32][33];
  int b = blockIdx.z;
  const float* ib = in + (size_t)b * R * S;
  __bf16* ob = out + (size_t)b * R * S;
  int x = blockIdx.x * 32 + threadIdx.x;  // S index
  int y0 = blockIdx.y * 32;               // R base
#pragma unroll
  for (int j = 0; j < 4; ++j)
    tile[threadIdx.y + j * 8][threadIdx.x] =
        ib[(size_t)(y0 + threadIdx.y + j * 8) * S + x];
  __syncthreads();
#pragma unroll
  for (int j = 0; j < 4; ++j) {
    int xo = blockIdx.x * 32 + threadIdx.y + j * 8;  // out row (S index)
    ob[(size_t)xo * R + y0 + threadIdx.x] =
        (__bf16)tile[threadIdx.x][threadIdx.y + j * 8];
  }
}

// ---------------------------------------------------------------------------
// 3) NT GEMM, bf16 out: out[1024,512] = A[1024,512].Bnt[512,512]^T per batch.
//    8 waves/block, each owns a 16-row strip (A preloaded, 128 VGPRs);
//    B tiles TDM-staged into LDS, double buffered, shared by all 8 waves.
// ---------------------------------------------------------------------------
__global__ __launch_bounds__(256) void gemm_nt_512_kernel(
    const __bf16* __restrict__ A, const __bf16* __restrict__ Bnt,
    __bf16* __restrict__ out) {
  const int KD = 512, ND = 512, NT = ND / 16;
  __shared__ __align__(128) __bf16 sB[2][LDS_TILE];

  int b = blockIdx.y;
  int tid = threadIdx.x;
  int lane = tid & 31, wv = tid >> 5;
  int strip = blockIdx.x * 8 + wv;  // 0..63
  int rA = lane & 15, h = lane >> 4;

  const __bf16* Ab = A + ((size_t)b * 1024 + strip * 16 + rA) * KD;
  const __bf16* Bb = Bnt + (size_t)b * ND * KD;
  __bf16* Ob = out + ((size_t)b * 1024 + strip * 16) * ND;

  v16bf av[16];
#pragma unroll
  for (int kb = 0; kb < 16; ++kb) av[kb] = load_a_frag(Ab, kb * 32 + h * 8);

  stage_issue(Bb, KD, sB[0], tid);
  for (int nt = 0; nt < NT; ++nt) {
    if (nt + 1 < NT) {
      stage_issue(Bb + (size_t)(nt + 1) * 16 * KD, KD, sB[(nt + 1) & 1], tid);
      stage_wait_pending(tid);
    } else {
      stage_wait_last(tid);
    }
    __syncthreads();
    const __bf16* Brow = sB[nt & 1] + rA * LDS_ROW + h * 16;
    v8f acc = {};
    K16_PIPELINED(acc, av, Brow);
#pragma unroll
    for (int v = 0; v < 8; ++v)
      Ob[(size_t)(v + 8 * h) * ND + nt * 16 + rA] = (__bf16)acc[v];
    __syncthreads();
  }
}

// ---------------------------------------------------------------------------
// 4) Softmax stats: stream S = A16.B16^T, online row max / sumexp.
// ---------------------------------------------------------------------------
__global__ __launch_bounds__(256) void attn_stats_kernel(
    const __bf16* __restrict__ A16, const __bf16* __restrict__ B16,
    float* __restrict__ rmaxO, float* __restrict__ rsumO) {
  const int KD = 512, NT = 64;
  __shared__ __align__(128) __bf16 sB[2][LDS_TILE];

  int b = blockIdx.y;
  int tid = threadIdx.x;
  int lane = tid & 31, wv = tid >> 5;
  int strip = blockIdx.x * 8 + wv;
  int rA = lane & 15, h = lane >> 4;

  const __bf16* Ab = A16 + ((size_t)b * 1024 + strip * 16 + rA) * KD;
  const __bf16* Bb = B16 + (size_t)b * 1024 * KD;

  v16bf av[16];
#pragma unroll
  for (int kb = 0; kb < 16; ++kb) av[kb] = load_a_frag(Ab, kb * 32 + h * 8);

  float rm[8], rs[8];
#pragma unroll
  for (int v = 0; v < 8; ++v) { rm[v] = -1.0e30f; rs[v] = 0.0f; }

  stage_issue(Bb, KD, sB[0], tid);
  for (int mt = 0; mt < NT; ++mt) {
    if (mt + 1 < NT) {
      stage_issue(Bb + (size_t)(mt + 1) * 16 * KD, KD, sB[(mt + 1) & 1], tid);
      stage_wait_pending(tid);
    } else {
      stage_wait_last(tid);
    }
    __syncthreads();
    const __bf16* Brow = sB[mt & 1] + rA * LDS_ROW + h * 16;
    v8f acc = {};
    K16_PIPELINED(acc, av, Brow);
#pragma unroll
    for (int v = 0; v < 8; ++v) {
      float t = acc[v];
      t = fmaxf(t, __shfl_xor(t, 1));
      t = fmaxf(t, __shfl_xor(t, 2));
      t = fmaxf(t, __shfl_xor(t, 4));
      t = fmaxf(t, __shfl_xor(t, 8));
      float nm = fmaxf(rm[v], t);
      float e = __builtin_amdgcn_exp2f((acc[v] - nm) * LOG2E);
      e += __shfl_xor(e, 1);
      e += __shfl_xor(e, 2);
      e += __shfl_xor(e, 4);
      e += __shfl_xor(e, 8);
      rs[v] = rs[v] * __builtin_amdgcn_exp2f((rm[v] - nm) * LOG2E) + e;
      rm[v] = nm;
    }
    __syncthreads();
  }
  if ((lane & 15) == 0) {
#pragma unroll
    for (int v = 0; v < 8; ++v) {
      size_t r = (size_t)b * 1024 + strip * 16 + v + 8 * h;
      rmaxO[r] = rm[v];
      rsumO[r] = rs[v];
    }
  }
}

// ---------------------------------------------------------------------------
// 5) Recompute S tiles, write normalized P (bf16) = exp(s-max)/sum.
// ---------------------------------------------------------------------------
__global__ __launch_bounds__(256) void attn_pwrite_kernel(
    const __bf16* __restrict__ A16, const __bf16* __restrict__ B16,
    const float* __restrict__ rmaxI, const float* __restrict__ rsumI,
    __bf16* __restrict__ P) {
  const int KD = 512, NT = 64;
  __shared__ __align__(128) __bf16 sB[2][LDS_TILE];

  int b = blockIdx.y;
  int tid = threadIdx.x;
  int lane = tid & 31, wv = tid >> 5;
  int strip = blockIdx.x * 8 + wv;
  int rA = lane & 15, h = lane >> 4;

  const __bf16* Ab = A16 + ((size_t)b * 1024 + strip * 16 + rA) * KD;
  const __bf16* Bb = B16 + (size_t)b * 1024 * KD;
  __bf16* Pb = P + ((size_t)b * 1024 + strip * 16) * 1024;

  v16bf av[16];
#pragma unroll
  for (int kb = 0; kb < 16; ++kb) av[kb] = load_a_frag(Ab, kb * 32 + h * 8);

  float mv[8], iv[8];
#pragma unroll
  for (int v = 0; v < 8; ++v) {
    size_t r = (size_t)b * 1024 + strip * 16 + v + 8 * h;
    mv[v] = rmaxI[r];
    iv[v] = __builtin_amdgcn_rcpf(rsumI[r]);
  }

  stage_issue(Bb, KD, sB[0], tid);
  for (int mt = 0; mt < NT; ++mt) {
    if (mt + 1 < NT) {
      stage_issue(Bb + (size_t)(mt + 1) * 16 * KD, KD, sB[(mt + 1) & 1], tid);
      stage_wait_pending(tid);
    } else {
      stage_wait_last(tid);
    }
    __syncthreads();
    const __bf16* Brow = sB[mt & 1] + rA * LDS_ROW + h * 16;
    v8f acc = {};
    K16_PIPELINED(acc, av, Brow);
#pragma unroll
    for (int v = 0; v < 8; ++v) {
      float p = __builtin_amdgcn_exp2f((acc[v] - mv[v]) * LOG2E) * iv[v];
      Pb[(size_t)(v + 8 * h) * 1024 + mt * 16 + rA] = (__bf16)p;
    }
    __syncthreads();
  }
}

// ---------------------------------------------------------------------------
// 6) Cout = P[1024,1024].Tt[512,1024]^T, += x_in, f32 store.
//    Wave owns 16 rows x 128 cols; K split in 2x512 chunks (A frag 128 VGPRs).
//    Tt tiles (16 rows x 1KB, global row stride 2KB) TDM-staged into LDS.
// ---------------------------------------------------------------------------
__global__ __launch_bounds__(256) void attn_out_kernel(
    const __bf16* __restrict__ P, const __bf16* __restrict__ Tt,
    const float* __restrict__ x_in, float* __restrict__ out) {
  __shared__ __align__(128) __bf16 sB[2][LDS_TILE];

  int b = blockIdx.z, cblk = blockIdx.y;
  int tid = threadIdx.x;
  int lane = tid & 31, wv = tid >> 5;
  int strip = blockIdx.x * 8 + wv;  // 0..63
  int rA = lane & 15, h = lane >> 4;
  int c0 = cblk * 128;

  const __bf16* Pb = P + ((size_t)b * 1024 + strip * 16 + rA) * 1024;
  const __bf16* Tb = Tt + (size_t)b * 512 * 1024;

  v8f acc[8];
#pragma unroll
  for (int nt = 0; nt < 8; ++nt) acc[nt] = (v8f){};

  // Stage s = kc_idx*8 + nt ; tile = Tt rows [c0+nt*16, +16), cols [kc, +512)
  stage_issue(Tb + (size_t)c0 * 1024, 1024, sB[0], tid);
  for (int kc_idx = 0; kc_idx < 2; ++kc_idx) {
    int kc = kc_idx * 512;
    v16bf av[16];
#pragma unroll
    for (int kb = 0; kb < 16; ++kb)
      av[kb] = load_a_frag(Pb + kc, kb * 32 + h * 8);
    for (int nt = 0; nt < 8; ++nt) {
      int s = kc_idx * 8 + nt;
      if (s + 1 < 16) {
        int nt2 = (s + 1) & 7, kc2 = ((s + 1) >> 3) * 512;
        stage_issue(Tb + (size_t)(c0 + nt2 * 16) * 1024 + kc2, 1024,
                    sB[(s + 1) & 1], tid);
        stage_wait_pending(tid);
      } else {
        stage_wait_last(tid);
      }
      __syncthreads();
      const __bf16* Brow = sB[s & 1] + rA * LDS_ROW + h * 16;
      K16_PIPELINED(acc[nt], av, Brow);
      __syncthreads();
    }
  }
#pragma unroll
  for (int nt = 0; nt < 8; ++nt) {
#pragma unroll
    for (int v = 0; v < 8; ++v) {
      size_t r = (size_t)strip * 16 + v + 8 * h;
      size_t idx = ((size_t)b * 1024 + r) * 512 + c0 + nt * 16 + rA;
      out[idx] = x_in[idx] + acc[nt][v];
    }
  }
}

// ---------------------------------------------------------------------------
// Host launch
// ---------------------------------------------------------------------------
extern "C" void kernel_launch(void* const* d_in, const int* in_sizes, int n_in,
                              void* d_out, int out_size, void* d_ws,
                              size_t ws_size, hipStream_t stream) {
  (void)in_sizes; (void)n_in; (void)out_size; (void)ws_size;
  const float* x_in = (const float*)d_in[0];  // [16,1024,512]
  const float* T    = (const float*)d_in[1];  // [16,1024,512]
  const float* Wq   = (const float*)d_in[2];  // [16,512,512]
  const float* Wk   = (const float*)d_in[3];  // [16,512,512]
  float* out = (float*)d_out;                 // [16,1024,512]

  const size_t MB = 1024ull * 1024ull;
  char* w = (char*)d_ws;
  __bf16* x16 = (__bf16*)(w + 0 * MB);     // 16 MB
  __bf16* T16 = (__bf16*)(w + 16 * MB);    // 16 MB
  __bf16* WqT = (__bf16*)(w + 32 * MB);    // 8 MB
  __bf16* WkT = (__bf16*)(w + 40 * MB);    // 8 MB
  __bf16* Tt  = (__bf16*)(w + 48 * MB);    // 16 MB
  __bf16* A16 = (__bf16*)(w + 64 * MB);    // 16 MB
  __bf16* B16 = (__bf16*)(w + 80 * MB);    // 16 MB
  __bf16* P   = (__bf16*)(w + 96 * MB);    // 32 MB
  float* rmax = (float*)(w + 128 * MB);    // 64 KB
  float* rsum = (float*)(w + 128 * MB + 64 * 1024);

  const int nXT = 16 * 1024 * 512;  // elements in x_in / T

  // Stage 1: bf16 staging (convert + transposed weights / T^T).
  cvt_bf16_kernel<<<nXT / 1024, 256, 0, stream>>>(x_in, x16, nXT);
  cvt_bf16_kernel<<<nXT / 1024, 256, 0, stream>>>(T, T16, nXT);
  transpose_cvt_kernel<<<dim3(16, 16, 16), dim3(32, 8), 0, stream>>>(Wq, WqT, 512, 512);
  transpose_cvt_kernel<<<dim3(16, 16, 16), dim3(32, 8), 0, stream>>>(Wk, WkT, 512, 512);
  transpose_cvt_kernel<<<dim3(16, 32, 16), dim3(32, 8), 0, stream>>>(T, Tt, 1024, 512);

  // Stage 2: projections A = x.Wq, Bm = T.Wk (bf16 WMMA, bf16 out).
  gemm_nt_512_kernel<<<dim3(8, 16), 256, 0, stream>>>(x16, WqT, A16);
  gemm_nt_512_kernel<<<dim3(8, 16), 256, 0, stream>>>(T16, WkT, B16);

  // Stage 3: softmax stats, then normalized P.
  attn_stats_kernel<<<dim3(8, 16), 256, 0, stream>>>(A16, B16, rmax, rsum);
  attn_pwrite_kernel<<<dim3(8, 16), 256, 0, stream>>>(A16, B16, rmax, rsum, P);

  // Stage 4: Cout = P.T, residual add, f32 out.
  attn_out_kernel<<<dim3(8, 4, 16), 256, 0, stream>>>(P, Tt, x_in, out);
}
